// MHA_41540923687542
// MI455X (gfx1250) — matrix operand order repaired and checked by reading, not compiled, not run on hardware
//
#include <hip/hip_runtime.h>
#include <hip/hip_bf16.h>
#include <math.h>

typedef __bf16 bf16_t;
typedef __attribute__((ext_vector_type(16))) __bf16        v16bf;
typedef __attribute__((ext_vector_type(8)))  __bf16        v8bf;
typedef __attribute__((ext_vector_type(8)))  float         v8f;
typedef __attribute__((ext_vector_type(4)))  unsigned int  u32v4;

#define DMODEL 256
#define BM 128
#define BN 128
#define BK 32
#define BKP 40   // padded LDS row stride (80 B: 16B-aligned, conflict-free bank stride)

union Frag16 { v16bf h; u32v4 q[2]; };

// ---------------------------------------------------------------------------
// Y[row0..+127, col0..+127] = X @ W + bias in bf16 WMMA, fp32 accumulate.
// 256 threads = 8 waves; wave w owns row-tile w (16 rows) x 8 col-tiles.
// K loop: 8 steps of 32.  All fragment loads are ds_load_b128; staging loops
// have fixed trip counts so no exec-mask churn.
// ---------------------------------------------------------------------------
template <bool IN_BF16, bool OUT_BF16>
__global__ __launch_bounds__(256)
void gemm_bias_bf16wmma(const void* __restrict__ Xv,
                        const float* __restrict__ W,
                        const float* __restrict__ bias,
                        void* __restrict__ Yv)
{
    __shared__ __align__(16) bf16_t As [BM * BKP];   // [row][k]   10 KB
    __shared__ __align__(16) bf16_t BsT[BN * BKP];   // [col][k]   10 KB

    const int tid  = threadIdx.x;
    const int wave = tid >> 5;
    const int lane = tid & 31;
    const int m    = lane & 15;
    const int half = lane >> 4;
    const int row0 = blockIdx.x * BM;
    const int col0 = blockIdx.y * BN;

    const float*  Xf = (const float*)Xv;
    const bf16_t* Xb = (const bf16_t*)Xv;

    v8f c[8];
#pragma unroll
    for (int t = 0; t < 8; ++t)
#pragma unroll
        for (int j = 0; j < 8; ++j) c[t][j] = 0.0f;

    for (int kk = 0; kk < DMODEL; kk += BK) {
        // ---- stage A tile: 128 rows x 32 k -> bf16 LDS [row][k] ----
        // 1024 quads / 256 threads = exactly 4 per thread (no bound checks)
#pragma unroll
        for (int it = 0; it < (BM * BK) / 4 / 256; ++it) {
            const int i  = tid + it * 256;
            const int r  = i >> 3;
            const int qd = i & 7;
            const int gr = row0 + r;
            const int gc = kk + qd * 4;
            bf16_t* dst = &As[r * BKP + qd * 4];
            if (IN_BF16) {
                const bf16_t* p = Xb + (size_t)gr * DMODEL + gc;
                dst[0] = p[0]; dst[1] = p[1]; dst[2] = p[2]; dst[3] = p[3];
                __builtin_prefetch(p + BK, 0, 1);   // speculative: OOB dropped
            } else {
                const float* p = Xf + (size_t)gr * DMODEL + gc;
                const float4 v = *(const float4*)p;
                dst[0] = (bf16_t)v.x; dst[1] = (bf16_t)v.y;
                dst[2] = (bf16_t)v.z; dst[3] = (bf16_t)v.w;
                __builtin_prefetch(p + BK, 0, 1);
            }
        }
        // ---- stage B tile: 32 k x 128 cols -> bf16 LDS transposed [col][k] ----
#pragma unroll
        for (int it = 0; it < (BK * BN) / 4 / 256; ++it) {
            const int i  = tid + it * 256;
            const int kr = i >> 5;          // k row (0..31)
            const int qd = i & 31;          // quad of columns
            const float4 v = *(const float4*)(W + (size_t)(kk + kr) * DMODEL + col0 + qd * 4);
            const int c0 = qd * 4;
            BsT[(c0 + 0) * BKP + kr] = (bf16_t)v.x;
            BsT[(c0 + 1) * BKP + kr] = (bf16_t)v.y;
            BsT[(c0 + 2) * BKP + kr] = (bf16_t)v.z;
            BsT[(c0 + 3) * BKP + kr] = (bf16_t)v.w;
        }
        __syncthreads();

        // ---- A fragment (ISA 7.12.2): lanes 0-15 K=0..7 & 16..23,
        //      lanes 16-31 K=8..15 & 24..31 -> two contiguous 16B chunks
        Frag16 a;
        const bf16_t* ar = &As[(wave * 16 + m) * BKP + half * 8];
        a.q[0] = *(const u32v4*)(ar);
        a.q[1] = *(const u32v4*)(ar + 16);

        // ---- 8 col tiles: B fragment = two ds_load_b128 from BsT ----
#pragma unroll
        for (int t = 0; t < 8; ++t) {
            Frag16 b;
            const bf16_t* br = &BsT[(t * 16 + m) * BKP + half * 8];
            b.q[0] = *(const u32v4*)(br);
            b.q[1] = *(const u32v4*)(br + 16);
            c[t] = __builtin_amdgcn_wmma_f32_16x16x32_bf16(
                false, a.h, false, b.h, (short)0, c[t], false, false);
        }
        __syncthreads();
    }

    // ---- epilogue: C/D layout — lane col = lane%16; VGPR v row = v + 8*(lane/16)
#pragma unroll
    for (int t = 0; t < 8; ++t) {
        const int n  = col0 + t * 16 + m;
        const float bv = bias[n];
#pragma unroll
        for (int v = 0; v < 8; ++v) {
            const int gm = row0 + wave * 16 + v + 8 * half;
            const float val = c[t][v] + bv;
            if (OUT_BF16)
                ((bf16_t*)Yv)[(size_t)gm * DMODEL + n] = (bf16_t)val;
            else
                ((float*)Yv)[(size_t)gm * DMODEL + n] = val;
        }
    }
}

// ---------------------------------------------------------------------------
// kNN attention: one wave32 per query. Lane = neighbour for E/softmax phase,
// lane = channel for the AV phase. bf16 data, fp32 accumulation. The gather
// working set (Kh+Vh bf16 = 16 MB) is L2-resident on MI455X (192 MB L2).
// ---------------------------------------------------------------------------
__global__ __launch_bounds__(256)
void knn_attn_kernel(const bf16_t* __restrict__ Qh,
                     const bf16_t* __restrict__ Kh,
                     const bf16_t* __restrict__ Vh,
                     const int*    __restrict__ neigh,
                     bf16_t*       __restrict__ AV,
                     int Nq, int Mrows)
{
    __shared__ __align__(16) float Qs[8][DMODEL];   // 8 KB: query row per wave
    __shared__ float Aw[8][8 * 32];                 // 8 KB: weights [head][k]
    __shared__ int   Idx[8][32];                    // 1 KB

    const int wave = threadIdx.x >> 5;
    const int lane = threadIdx.x & 31;
    const int q    = blockIdx.x * 8 + wave;         // global query in [0, B*N)
    const int b    = q / Nq;

    // stage query row (bf16 -> fp32), vectorized
    const v8bf* qv8 = (const v8bf*)(Qh + (size_t)q * DMODEL);
    {
        const v8bf qq = qv8[lane];
#pragma unroll
        for (int j = 0; j < 8; ++j) Qs[wave][lane * 8 + j] = (float)qq[j];
    }
    const int mi = neigh[(size_t)q * 32 + lane];
    Idx[wave][lane] = mi;
    __syncthreads();

    // phase 1: lane = neighbour; per-head dot products (b128 loads both sides)
    const bf16_t* krow = Kh + ((size_t)b * Mrows + mi) * DMODEL;
    float e[8];
#pragma unroll
    for (int h = 0; h < 8; ++h) {
        const v8bf*   kv = (const v8bf*)(krow + h * 32);
        const float4* qv = (const float4*)&Qs[wave][h * 32];
        float acc = 0.0f;
#pragma unroll
        for (int g = 0; g < 4; ++g) {
            const v8bf   k8 = kv[g];
            const float4 q0 = qv[2 * g + 0];
            const float4 q1 = qv[2 * g + 1];
            acc += q0.x * (float)k8[0] + q0.y * (float)k8[1]
                 + q0.z * (float)k8[2] + q0.w * (float)k8[3]
                 + q1.x * (float)k8[4] + q1.y * (float)k8[5]
                 + q1.z * (float)k8[6] + q1.w * (float)k8[7];
        }
        e[h] = acc * 0.0625f;   // 1/sqrt(dv=256), faithful to reference
    }

    // softmax over the 32 neighbours (cross-lane, wave32-native)
#pragma unroll
    for (int h = 0; h < 8; ++h) {
        float mx = e[h];
#pragma unroll
        for (int s = 16; s > 0; s >>= 1) mx = fmaxf(mx, __shfl_xor(mx, s, 32));
        const float p = __expf(e[h] - mx);
        float sm = p;
#pragma unroll
        for (int s = 16; s > 0; s >>= 1) sm += __shfl_xor(sm, s, 32);
        Aw[wave][h * 32 + lane] = p / sm;
    }
    __syncthreads();

    // phase 2: lane = channel (d = j*32 + lane, head = j); AV = sum_k A * Vn
    float out[8];
#pragma unroll
    for (int j = 0; j < 8; ++j) out[j] = 0.0f;
    for (int kk = 0; kk < 32; ++kk) {
        const int mk = Idx[wave][kk];
        const bf16_t* vrow = Vh + ((size_t)b * Mrows + mk) * DMODEL;
#pragma unroll
        for (int j = 0; j < 8; ++j)
            out[j] += Aw[wave][j * 32 + kk] * (float)vrow[j * 32 + lane];
    }
    bf16_t* av = AV + (size_t)q * DMODEL;
#pragma unroll
    for (int j = 0; j < 8; ++j) av[j * 32 + lane] = (bf16_t)out[j];
}

// ---------------------------------------------------------------------------
extern "C" void kernel_launch(void* const* d_in, const int* in_sizes, int n_in,
                              void* d_out, int out_size, void* d_ws, size_t ws_size,
                              hipStream_t stream)
{
    const float* Q     = (const float*)d_in[0];
    const float* K     = (const float*)d_in[1];
    const int*   neigh = (const int*)  d_in[2];
    const float* Wq    = (const float*)d_in[3];
    const float* bq    = (const float*)d_in[4];
    const float* Wk    = (const float*)d_in[5];
    const float* bk    = (const float*)d_in[6];
    const float* Wv    = (const float*)d_in[7];
    const float* bv    = (const float*)d_in[8];
    const float* Wo    = (const float*)d_in[9];
    const float* bo    = (const float*)d_in[10];

    const int B = 2, N = 8192, M = 8192;
    const size_t rowsQ = (size_t)B * N;   // 16384
    const size_t rowsK = (size_t)B * M;   // 16384

    // workspace: Qh | Kh | Vh | AV, all bf16 [rows, 256] -> 32 MB total
    bf16_t* Qh = (bf16_t*)d_ws;
    bf16_t* Kh = Qh + rowsQ * DMODEL;
    bf16_t* Vh = Kh + rowsK * DMODEL;
    bf16_t* AV = Vh + rowsK * DMODEL;

    const dim3 blk(256);
    const dim3 gq((unsigned)(rowsQ / BM), DMODEL / BN);
    const dim3 gk((unsigned)(rowsK / BM), DMODEL / BN);

    gemm_bias_bf16wmma<false, true><<<gq, blk, 0, stream>>>(Q, Wq, bq, Qh);
    gemm_bias_bf16wmma<false, true><<<gk, blk, 0, stream>>>(K, Wk, bk, Kh);
    gemm_bias_bf16wmma<false, true><<<gk, blk, 0, stream>>>(K, Wv, bv, Vh);

    knn_attn_kernel<<<(unsigned)(rowsQ / 8), blk, 0, stream>>>(Qh, Kh, Vh, neigh, AV, N, M);

    gemm_bias_bf16wmma<true, false><<<gq, blk, 0, stream>>>(AV, Wo, bo, (float*)d_out);
}
